// GCGAT_v4pro_16604343566710
// MI455X (gfx1250) — compile-verified, matrix operand order checked by reading, not run on hardware
//
#include <hip/hip_runtime.h>
#include <hip/hip_bf16.h>
#include <cstdint>
#include <cstddef>

// ---------------- problem constants (match reference) ----------------
constexpr int N_ATOM = 131072, E_ATOM = 262144;
constexpr int N_FRAG = 131072, E_FRAG = 196608;
constexpr int N_JUNC = 32768,  E_JUNC = 49152;
constexpr int NGRAPH = 8192;
constexpr int H = 128, DIN = 44, EIN = 12, FRAG_DIM = 219;
constexpr int H3 = 3 * H;
constexpr int HLOG = 7;           // log2(H)

enum { ACT_NONE = 0, ACT_LRELU = 1, ACT_RELU = 2, ACT_ELU = 3, ACT_LRELU7 = 4 };

typedef __attribute__((ext_vector_type(16))) __bf16 v16bf;
typedef __attribute__((ext_vector_type(8)))  __bf16 v8bf;
typedef __attribute__((ext_vector_type(8)))  float  v8f;

#define DIV_UP(a, b) (((a) + (b) - 1) / (b))

// ---------------- device helpers ----------------
__device__ __forceinline__ unsigned pack_bf16x2(float a, float b) {
    // packed RNE f32->bf16x2; lowers to v_cvt_pk_bf16_f32 when available
    __hip_bfloat162 h = __float22bfloat162_rn(float2{a, b});
    unsigned u;
    __builtin_memcpy(&u, &h, 4);
    return u;
}
__device__ __forceinline__ float apply_act(float v, int act) {
    switch (act) {
        case ACT_LRELU:  return v >= 0.f ? v : 0.01f * v;
        case ACT_RELU:   return v > 0.f ? v : 0.f;
        case ACT_ELU:    return v > 0.f ? v : (__expf(v) - 1.f);
        case ACT_LRELU7: return v >= 0.f ? v : 1e-7f * v;
    }
    return v;
}
__device__ __forceinline__ unsigned f2ord(float f) {
    unsigned u = __float_as_uint(f);
    return (u & 0x80000000u) ? ~u : (u | 0x80000000u);
}
__device__ __forceinline__ float ord2f(unsigned u) {
    return __uint_as_float((u & 0x80000000u) ? (u & 0x7FFFFFFFu) : ~u);
}
__device__ __forceinline__ float wave_reduce(float v) {
    #pragma unroll
    for (int o = 16; o; o >>= 1) v += __shfl_xor(v, o, 32);
    return v;
}
// K-column permutation so each lane's WMMA fragment is 16 contiguous bf16 in LDS.
// ISA frag order per lane: K in {khi..khi+7} then {16+khi..16+khi+7}, khi = 8*(lane>=16).
// Permuted column order [0..7,16..23,8..15,24..31]: swap 8-groups 1 and 2.
__device__ __forceinline__ int kperm(int k) {
    int g = k >> 3;
    g = ((g & 1) << 1) | (g >> 1);
    return (g << 3) | (k & 7);
}

// ---------------- WMMA GEMM: C[M,N] = act(Agather @ W + bias (+C)) ----------------
// A fp32 [M,K] (row-gathered iff GATHER), W fp32 [K,N]; bf16 WMMA 16x16x32, f32 acc.
// REQUIRES (guaranteed by all call sites): M % 128 == 0, N % 64 == 0. K arbitrary.
// Tile 128x64 per 256-thread block (8 waves); wave w -> rows [16w,16w+16), all 64 cols.
// Double-buffered LDS, fragment-order layout: each fragment = two aligned ds_load_b128,
// row stride 20 dwords -> conflict-free across the 16 lanes of each half-wave.
template<bool GATHER>
__global__ __launch_bounds__(256)
void k_gemm_t(const float* __restrict__ A, const float* __restrict__ W,
              const float* __restrict__ bias, float* __restrict__ C,
              int M, int N, int K, int ldc,
              const int* __restrict__ gatherA, int act, int accum)
{
    (void)M;
    __shared__ __align__(16) __bf16 As[2][128][40];   // 2 x 10240 B
    __shared__ __align__(16) __bf16 Bs[2][64][40];    // 2 x  5120 B

    const int tid = threadIdx.x, wv = tid >> 5, lane = tid & 31;
    const int m0 = blockIdx.y * 128, n0 = blockIdx.x * 64;

    v8f acc[4];
    #pragma unroll
    for (int nb = 0; nb < 4; ++nb)
        #pragma unroll
        for (int i = 0; i < 8; ++i) acc[nb][i] = 0.f;

    // ---- per-thread fill coordinates & source pointers (hoisted out of K loop) ----
    const float* asrc[8]; int a_r[8], a_c[8];
    #pragma unroll
    for (int i = 0; i < 8; ++i) {
        int p = tid + i * 256;
        a_r[i] = p >> 4;                 // 0..127 (tile M)
        a_c[i] = (p & 15) << 1;          // even k within tile
        int row = m0 + a_r[i];
        size_t ar = GATHER ? (size_t)gatherA[row] : (size_t)row;
        asrc[i] = A + ar * K + a_c[i];
    }
    const float* bsrc[4]; int b_c[4], b_r[4];
    #pragma unroll
    for (int i = 0; i < 4; ++i) {
        int p = tid + i * 256;
        b_c[i] = p >> 4;                 // 0..63 (tile N)
        b_r[i] = (p & 15) << 1;          // even k within tile
        bsrc[i] = W + (size_t)b_r[i] * N + (n0 + b_c[i]);
    }

    float a0[8], a1[8], b0[4], b1[4];
    auto load_tile = [&](int k0) {
        if (k0 + 32 <= K) {              // fast path: no per-element guards
            #pragma unroll
            for (int i = 0; i < 8; ++i) { a0[i] = asrc[i][k0]; a1[i] = asrc[i][k0 + 1]; }
            #pragma unroll
            for (int i = 0; i < 4; ++i) {
                b0[i] = bsrc[i][(size_t)k0 * N];
                b1[i] = bsrc[i][(size_t)(k0 + 1) * N];
            }
        } else {                         // K-tail tile only
            #pragma unroll
            for (int i = 0; i < 8; ++i) {
                a0[i] = (k0 + a_c[i]     < K) ? asrc[i][k0]     : 0.f;
                a1[i] = (k0 + a_c[i] + 1 < K) ? asrc[i][k0 + 1] : 0.f;
            }
            #pragma unroll
            for (int i = 0; i < 4; ++i) {
                b0[i] = (k0 + b_r[i]     < K) ? bsrc[i][(size_t)k0 * N]       : 0.f;
                b1[i] = (k0 + b_r[i] + 1 < K) ? bsrc[i][(size_t)(k0 + 1) * N] : 0.f;
            }
        }
    };
    auto store_tile = [&](int buf) {
        #pragma unroll
        for (int i = 0; i < 8; ++i)
            *(unsigned*)(&As[buf][a_r[i]][kperm(a_c[i])]) = pack_bf16x2(a0[i], a1[i]);
        #pragma unroll
        for (int i = 0; i < 4; ++i)
            *(unsigned*)(&Bs[buf][b_c[i]][kperm(b_r[i])]) = pack_bf16x2(b0[i], b1[i]);
    };

    // per-lane fragment source offsets (constant across K loop)
    const int a_off = (wv * 16 + (lane & 15)) * 40 + (lane >> 4) * 16;
    const int b_off = (lane & 15) * 40 + (lane >> 4) * 16;
    const __bf16* Abase = &As[0][0][0];
    const __bf16* Bbase = &Bs[0][0][0];

    const int ntiles = DIV_UP(K, 32);
    load_tile(0);
    store_tile(0);
    for (int t = 0; t < ntiles; ++t) {
        __syncthreads();
        if (t + 1 < ntiles) {
            load_tile((t + 1) * 32);     // issue global loads; latency hides under WMMA
            if (t + 2 < ntiles) {
                __builtin_prefetch(&asrc[0][(t + 2) * 32], 0, 1);
                __builtin_prefetch(&bsrc[0][(size_t)(t + 2) * 32 * N], 0, 1);
            }
        }
        // load A frag + all 4 B frags into distinct registers first, so the
        // ds_load_b128s pipeline and the 4 WMMAs issue back-to-back.
        const __bf16* ap  = Abase + (t & 1) * (128 * 40) + a_off;
        const __bf16* bp0 = Bbase + (t & 1) * (64 * 40) + b_off;
        v8bf alo = *(const v8bf*)ap;
        v8bf ahi = *(const v8bf*)(ap + 8);
        v16bf afrag = __builtin_shufflevector(alo, ahi,
            0, 1, 2, 3, 4, 5, 6, 7, 8, 9, 10, 11, 12, 13, 14, 15);
        v16bf bfrag[4];
        #pragma unroll
        for (int nb = 0; nb < 4; ++nb) {
            const __bf16* bp = bp0 + nb * 16 * 40;
            v8bf blo = *(const v8bf*)bp;
            v8bf bhi = *(const v8bf*)(bp + 8);
            bfrag[nb] = __builtin_shufflevector(blo, bhi,
                0, 1, 2, 3, 4, 5, 6, 7, 8, 9, 10, 11, 12, 13, 14, 15);
        }
        #pragma unroll
        for (int nb = 0; nb < 4; ++nb)
            acc[nb] = __builtin_amdgcn_wmma_f32_16x16x32_bf16(
                false, afrag, false, bfrag[nb], (short)0, acc[nb], false, false);
        if (t + 1 < ntiles) store_tile((t + 1) & 1);
    }

    // ---- epilogue (no M/N guards: tiles are exact) ----
    const int rbase = m0 + wv * 16 + ((lane >> 4) << 3);
    const int cl = lane & 15;
    #pragma unroll
    for (int nb = 0; nb < 4; ++nb) {
        int col = n0 + nb * 16 + cl;
        float bv = bias ? bias[col] : 0.f;
        #pragma unroll
        for (int i = 0; i < 8; ++i) {
            float v = acc[nb][i] + bv;
            size_t idx = (size_t)(rbase + i) * ldc + col;
            if (accum) v += C[idx];
            C[idx] = apply_act(v, act);
        }
    }
}

// ---------------- streaming / segment kernels ----------------
__global__ void k_fill_f32(float* p, float v, size_t n) {
    for (size_t t = (size_t)blockIdx.x * blockDim.x + threadIdx.x; t < n;
         t += (size_t)gridDim.x * blockDim.x) p[t] = v;
}
__global__ void k_fill_u32(unsigned* p, unsigned v, size_t n) {
    for (size_t t = (size_t)blockIdx.x * blockDim.x + threadIdx.x; t < n;
         t += (size_t)gridDim.x * blockDim.x) p[t] = v;
}

// per-edge attention score: s = lrelu(h[src].a_src + h[dst].a_dst + he.a_e)
__global__ __launch_bounds__(256)
void k_edge_att(const float* __restrict__ h, const float* __restrict__ he,
                const int* __restrict__ src, const int* __restrict__ dst,
                const float* __restrict__ a_src, const float* __restrict__ a_dst,
                const float* __restrict__ a_e, float* __restrict__ s, int E)
{
    int e = blockIdx.x * (blockDim.x >> 5) + (threadIdx.x >> 5);
    int lane = threadIdx.x & 31;
    if (e >= E) return;
    int si = src[e], di = dst[e];
    float acc = 0.f;
    #pragma unroll
    for (int c = lane; c < H; c += 32)
        acc += h[((size_t)si << HLOG) + c] * a_src[c]
             + h[((size_t)di << HLOG) + c] * a_dst[c]
             + he[((size_t)e << HLOG) + c] * a_e[c];
    acc = wave_reduce(acc);
    if (lane == 0) s[e] = acc >= 0.f ? acc : 0.01f * acc;
}

// per-node readout score: s = lrelu(g[batch].a_g + h.a_h)
__global__ __launch_bounds__(256)
void k_mol_att(const float* __restrict__ g, const float* __restrict__ h,
               const int* __restrict__ batch, const float* __restrict__ a_g,
               const float* __restrict__ a_h, float* __restrict__ s, int N)
{
    int n = blockIdx.x * (blockDim.x >> 5) + (threadIdx.x >> 5);
    int lane = threadIdx.x & 31;
    if (n >= N) return;
    int b = batch[n];
    float acc = 0.f;
    #pragma unroll
    for (int c = lane; c < H; c += 32)
        acc += g[((size_t)b << HLOG) + c] * a_g[c] + h[((size_t)n << HLOG) + c] * a_h[c];
    acc = wave_reduce(acc);
    if (lane == 0) s[n] = acc >= 0.f ? acc : 0.01f * acc;
}

__global__ void k_segmax(const float* __restrict__ s, const int* __restrict__ seg,
                         unsigned* __restrict__ mx, int n) {
    for (size_t t = (size_t)blockIdx.x * blockDim.x + threadIdx.x; t < (size_t)n;
         t += (size_t)gridDim.x * blockDim.x)
        atomicMax(&mx[seg[t]], f2ord(s[t]));
}
__global__ void k_segexp(const float* __restrict__ s, const int* __restrict__ seg,
                         const unsigned* __restrict__ mx, float* __restrict__ den,
                         float* __restrict__ ev, int n) {
    for (size_t t = (size_t)blockIdx.x * blockDim.x + threadIdx.x; t < (size_t)n;
         t += (size_t)gridDim.x * blockDim.x) {
        float e = __expf(s[t] - ord2f(mx[seg[t]]));
        ev[t] = e;
        atomicAdd(&den[seg[t]], e);
    }
}
// agg[seg] += alpha * msg   (alpha = ev/den per edge/node)
__global__ void k_scatter(const float* __restrict__ msg, const float* __restrict__ ev,
                          const float* __restrict__ den, const int* __restrict__ seg,
                          float* __restrict__ agg, size_t total) {
    for (size_t t = (size_t)blockIdx.x * blockDim.x + threadIdx.x; t < total;
         t += (size_t)gridDim.x * blockDim.x) {
        size_t e = t >> HLOG;
        int c = (int)(t & (H - 1));
        int d = seg[e];
        float a = ev[e] / (den[d] + 1e-16f);
        atomicAdd(&agg[((size_t)d << HLOG) + c], a * msg[t]);
    }
}
__global__ void k_segsum_feat(const float* __restrict__ x, const int* __restrict__ seg,
                              float* __restrict__ out, size_t rows) {
    size_t total = rows << HLOG;
    for (size_t t = (size_t)blockIdx.x * blockDim.x + threadIdx.x; t < total;
         t += (size_t)gridDim.x * blockDim.x) {
        size_t r = t >> HLOG;
        int c = (int)(t & (H - 1));
        atomicAdd(&out[((size_t)seg[r] << HLOG) + c], x[t]);
    }
}
__global__ void k_gru(const float* __restrict__ gi, const float* __restrict__ gh,
                      float* __restrict__ h, size_t rows) {
    size_t total = rows << HLOG;
    for (size_t t = (size_t)blockIdx.x * blockDim.x + threadIdx.x; t < total;
         t += (size_t)gridDim.x * blockDim.x) {
        size_t r = t >> HLOG;
        int c = (int)(t & (H - 1));
        size_t b3 = r * H3;
        float ir = gi[b3 + c], iz = gi[b3 + H + c], in = gi[b3 + 2 * H + c];
        float hr = gh[b3 + c], hz = gh[b3 + H + c], hn = gh[b3 + 2 * H + c];
        float rr = 1.f / (1.f + __expf(-(ir + hr)));
        float z  = 1.f / (1.f + __expf(-(iz + hz)));
        float nn = tanhf(in + rr * hn);
        h[t] = (1.f - z) * nn + z * h[t];
    }
}
__global__ void k_elu(float* p, size_t n) {
    for (size_t t = (size_t)blockIdx.x * blockDim.x + threadIdx.x; t < n;
         t += (size_t)gridDim.x * blockDim.x) {
        float v = p[t];
        p[t] = v > 0.f ? v : (__expf(v) - 1.f);
    }
}
__global__ void k_colstats(const float* __restrict__ x, size_t M, int C,
                           float* __restrict__ s1, float* __restrict__ s2) {
    size_t total = M * C;
    for (size_t t = (size_t)blockIdx.x * blockDim.x + threadIdx.x; t < total;
         t += (size_t)gridDim.x * blockDim.x) {
        int c = (int)(t % C);
        float v = x[t];
        atomicAdd(&s1[c], v);
        atomicAdd(&s2[c], v * v);
    }
}
__global__ void k_bn_apply(const float* __restrict__ x, const float* __restrict__ s1,
                           const float* __restrict__ s2, const float* __restrict__ gamma,
                           const float* __restrict__ beta, float* __restrict__ out,
                           size_t M, int C, int act) {
    size_t total = M * C;
    float invM = 1.f / (float)M;
    for (size_t t = (size_t)blockIdx.x * blockDim.x + threadIdx.x; t < total;
         t += (size_t)gridDim.x * blockDim.x) {
        int c = (int)(t % C);
        float mean = s1[c] * invM;
        float var  = s2[c] * invM - mean * mean;
        float v = (x[t] - mean) * rsqrtf(var + 1e-5f) * gamma[c] + beta[c];
        out[t] = apply_act(v, act);
    }
}
__global__ void k_copy_cols(const float* __restrict__ src, size_t M, int C,
                            float* __restrict__ dst, int dld) {
    size_t total = M * C;
    for (size_t t = (size_t)blockIdx.x * blockDim.x + threadIdx.x; t < total;
         t += (size_t)gridDim.x * blockDim.x) {
        size_t r = t / C;
        int c = (int)(t - r * C);
        dst[r * dld + c] = src[t];
    }
}
__global__ void k_mean2relu(const float* a, const float* b, float* out, size_t n) {
    for (size_t t = (size_t)blockIdx.x * blockDim.x + threadIdx.x; t < n;
         t += (size_t)gridDim.x * blockDim.x) {
        float v = 0.5f * (a[t] + b[t]);
        out[t] = v > 0.f ? v : 0.f;
    }
}
__global__ __launch_bounds__(256)
void k_gemv_out(const float* __restrict__ X, const float* __restrict__ Wv,
                const float* __restrict__ b, float* __restrict__ out, int M, int K) {
    int row = blockIdx.x * (blockDim.x >> 5) + (threadIdx.x >> 5);
    int lane = threadIdx.x & 31;
    if (row >= M) return;
    float acc = 0.f;
    for (int k = lane; k < K; k += 32) acc += X[(size_t)row * K + k] * Wv[k];
    acc = wave_reduce(acc);
    if (lane == 0) {
        float v = acc + b[0];
        out[row] = v >= 0.f ? v : 1e-7f * v;
    }
}

// ---------------- host-side parameter decoding (JAX alphabetical flatten) ----------------
struct GruP  { const float *Wh, *Wi, *bh, *bi; };
struct LayerP{ const float *Wm_e, *Wm_src, *a_dst, *a_e, *a_src, *bm; GruP gru; };
struct MolP  { const float *Wm, *a_g, *a_h; GruP gru; };
struct AfpP  { const float *We, *Wn, *Wo; LayerP lay[3]; const float *be, *bn, *bo; MolP mol; };
struct ChanP { const float *att_W, *att_b, *att_beta, *att_g; AfpP head[2]; };
struct JHeadP{ AfpP afp; const float *proj_W, *proj_b; };
struct JuncP { const float *emb_eW, *emb_eb, *emb_ebeta, *emb_eg;
               const float *emb_fW, *emb_fb, *emb_fbeta, *emb_fg; JHeadP head[2]; };

static inline const float* F(void* const* d, int& i) { return (const float*)d[i++]; }
static void read_gru(void* const* d, int& i, GruP& g) { g.Wh=F(d,i); g.Wi=F(d,i); g.bh=F(d,i); g.bi=F(d,i); }
static void read_afp(void* const* d, int& i, AfpP& p) {
    p.We=F(d,i); p.Wn=F(d,i); p.Wo=F(d,i);
    for (int l = 0; l < 3; ++l) {
        LayerP& L = p.lay[l];
        L.Wm_e=F(d,i); L.Wm_src=F(d,i); L.a_dst=F(d,i); L.a_e=F(d,i); L.a_src=F(d,i); L.bm=F(d,i);
        read_gru(d,i,L.gru);
    }
    p.be=F(d,i); p.bn=F(d,i); p.bo=F(d,i);
    p.mol.Wm=F(d,i); p.mol.a_g=F(d,i); p.mol.a_h=F(d,i); read_gru(d,i,p.mol.gru);
}
static void read_chan(void* const* d, int& i, ChanP& c) {
    c.att_W=F(d,i); c.att_b=F(d,i); c.att_beta=F(d,i); c.att_g=F(d,i);
    read_afp(d,i,c.head[0]); read_afp(d,i,c.head[1]);
}
static void read_junc(void* const* d, int& i, JuncP& j) {
    j.emb_eW=F(d,i); j.emb_eb=F(d,i); j.emb_ebeta=F(d,i); j.emb_eg=F(d,i);
    j.emb_fW=F(d,i); j.emb_fb=F(d,i); j.emb_fbeta=F(d,i); j.emb_fg=F(d,i);
    for (int h = 0; h < 2; ++h) { read_afp(d,i,j.head[h].afp); j.head[h].proj_W=F(d,i); j.head[h].proj_b=F(d,i); }
}

// ---------------- orchestration ----------------
struct GraphDesc { const float* x; int din; const float* ea; int ein;
                   const int *src, *dst, *batch; int Nn, Ne, Ng; };
struct WS {
    float *h, *he, *agg, *big1, *big2;     // big pool (atom-sized)
    float *s, *expv; unsigned *segmax; float *segden;
    float *g, *ctx, *xe, *ee, *xcat, *xh, *oc;
    float *graph_origin, *graph_frag, *frag_pool, *superg, *concat3, *d1, *d2;
    float *colsum, *colsumsq;
};

static inline int gsb(size_t n) { size_t b = (n + 255) / 256; return (int)(b > 8192 ? 8192 : b); }
static void gemm(hipStream_t st, const float* A, const float* W, const float* bias, float* C,
                 int M, int N, int K, int ldc, const int* gather, int act, int accum) {
    dim3 grid(DIV_UP(N, 64), DIV_UP(M, 128));
    if (gather)
        k_gemm_t<true><<<grid, 256, 0, st>>>(A, W, bias, C, M, N, K, ldc, gather, act, accum);
    else
        k_gemm_t<false><<<grid, 256, 0, st>>>(A, W, bias, C, M, N, K, ldc, nullptr, act, accum);
}
static void fillf(hipStream_t st, float* p, float v, size_t n) { k_fill_f32<<<gsb(n),256,0,st>>>(p,v,n); }
static void fillu(hipStream_t st, unsigned* p, unsigned v, size_t n) { k_fill_u32<<<gsb(n),256,0,st>>>(p,v,n); }
static void bn_block(hipStream_t st, WS& w, const float* x, size_t M, int C,
                     const float* gamma, const float* beta, float* out, int act) {
    fillf(st, w.colsum, 0.f, C); fillf(st, w.colsumsq, 0.f, C);
    k_colstats<<<gsb(M * C), 256, 0, st>>>(x, M, C, w.colsum, w.colsumsq);
    k_bn_apply<<<gsb(M * C), 256, 0, st>>>(x, w.colsum, w.colsumsq, gamma, beta, out, M, C, act);
}

static void run_afp(hipStream_t st, const AfpP& P, const GraphDesc& G, WS& w,
                    float* out, int out_ld) {
    const int Nn = G.Nn, Ne = G.Ne, Ng = G.Ng;
    // node / edge embeddings
    gemm(st, G.x,  P.Wn, P.bn, w.h,  Nn, H, G.din, H, nullptr, ACT_LRELU, 0);
    gemm(st, G.ea, P.We, P.be, w.he, Ne, H, G.ein, H, nullptr, ACT_LRELU, 0);
    for (int l = 0; l < 3; ++l) {
        const LayerP& L = P.lay[l];
        k_edge_att<<<DIV_UP(Ne, 8), 256, 0, st>>>(w.h, w.he, G.src, G.dst,
                                                  L.a_src, L.a_dst, L.a_e, w.s, Ne);
        fillu(st, w.segmax, 0u, Nn);
        k_segmax<<<gsb(Ne), 256, 0, st>>>(w.s, G.dst, w.segmax, Ne);
        fillf(st, w.segden, 0.f, Nn);
        k_segexp<<<gsb(Ne), 256, 0, st>>>(w.s, G.dst, w.segmax, w.segden, w.expv, Ne);
        // msg = elu(h[src]@Wm_src + he@Wm_e + bm) -> big1
        gemm(st, w.h,  L.Wm_src, L.bm,   w.big1, Ne, H, H, H, G.src,  ACT_NONE, 0);
        gemm(st, w.he, L.Wm_e,   nullptr,w.big1, Ne, H, H, H, nullptr, ACT_ELU, 1);
        fillf(st, w.agg, 0.f, (size_t)Nn << HLOG);
        k_scatter<<<gsb((size_t)Ne << HLOG), 256, 0, st>>>(w.big1, w.expv, w.segden, G.dst,
                                                           w.agg, (size_t)Ne << HLOG);
        // GRU: gi -> big1 (msg dead), gh -> big2
        gemm(st, w.agg, L.gru.Wi, L.gru.bi, w.big1, Nn, H3, H, H3, nullptr, ACT_NONE, 0);
        gemm(st, w.h,   L.gru.Wh, L.gru.bh, w.big2, Nn, H3, H, H3, nullptr, ACT_NONE, 0);
        k_gru<<<gsb((size_t)Nn << HLOG), 256, 0, st>>>(w.big1, w.big2, w.h, Nn);
    }
    // readout
    fillf(st, w.g, 0.f, (size_t)Ng << HLOG);
    k_segsum_feat<<<gsb((size_t)Nn << HLOG), 256, 0, st>>>(w.h, G.batch, w.g, Nn);
    gemm(st, w.h, P.mol.Wm, nullptr, w.big1, Nn, H, H, H, nullptr, ACT_NONE, 0);  // h@Wm (loop-invariant)
    float* gi_g = w.big2;
    float* gh_g = w.big2 + (size_t)Ng * H3;
    for (int it = 0; it < 2; ++it) {
        k_mol_att<<<DIV_UP(Nn, 8), 256, 0, st>>>(w.g, w.h, G.batch, P.mol.a_g, P.mol.a_h, w.s, Nn);
        fillu(st, w.segmax, 0u, Ng);
        k_segmax<<<gsb(Nn), 256, 0, st>>>(w.s, G.batch, w.segmax, Nn);
        fillf(st, w.segden, 0.f, Ng);
        k_segexp<<<gsb(Nn), 256, 0, st>>>(w.s, G.batch, w.segmax, w.segden, w.expv, Nn);
        fillf(st, w.ctx, 0.f, (size_t)Ng << HLOG);
        k_scatter<<<gsb((size_t)Nn << HLOG), 256, 0, st>>>(w.big1, w.expv, w.segden, G.batch,
                                                           w.ctx, (size_t)Nn << HLOG);
        k_elu<<<gsb((size_t)Ng << HLOG), 256, 0, st>>>(w.ctx, (size_t)Ng << HLOG);
        gemm(st, w.ctx, P.mol.gru.Wi, P.mol.gru.bi, gi_g, Ng, H3, H, H3, nullptr, ACT_NONE, 0);
        gemm(st, w.g,   P.mol.gru.Wh, P.mol.gru.bh, gh_g, Ng, H3, H, H3, nullptr, ACT_NONE, 0);
        k_gru<<<gsb((size_t)Ng << HLOG), 256, 0, st>>>(gi_g, gh_g, w.g, Ng);
    }
    gemm(st, w.g, P.Wo, P.bo, out, Ng, H, H, out_ld, nullptr, ACT_NONE, 0);
}

extern "C" void kernel_launch(void* const* d_in, const int* in_sizes, int n_in,
                              void* d_out, int out_size, void* d_ws, size_t ws_size,
                              hipStream_t stream) {
    (void)in_sizes; (void)n_in; (void)out_size; (void)ws_size;
    const float* x_atom  = (const float*)d_in[0];
    const float* ea_atom = (const float*)d_in[1];
    const float* x_frag  = (const float*)d_in[2];
    const float* ea_frag = (const float*)d_in[3];
    const float* x_motif = (const float*)d_in[4];
    const float* ea_junc = (const float*)d_in[5];
    int pi = 6;
    ChanP fragP; read_chan(d_in, pi, fragP);
    JuncP juncP; read_junc(d_in, pi, juncP);
    ChanP origP; read_chan(d_in, pi, origP);
    const float* p1W = F(d_in, pi); const float* p1b = F(d_in, pi);
    const float* p1beta = F(d_in, pi); const float* p1g = F(d_in, pi);
    const float* p2W = F(d_in, pi); const float* p2b = F(d_in, pi);
    const float* poW = F(d_in, pi); const float* pob = F(d_in, pi);
    const int* ei_a = (const int*)d_in[pi++]; const int* b_a = (const int*)d_in[pi++];
    const int* ei_f = (const int*)d_in[pi++]; const int* b_f = (const int*)d_in[pi++];
    const int* ei_j = (const int*)d_in[pi++]; const int* b_j = (const int*)d_in[pi++];

    // ---- workspace bump allocator (~880 MB peak) ----
    char* base = (char*)d_ws; size_t off = 0;
    auto alloc = [&](size_t nfloats) -> float* {
        float* p = (float*)(base + off);
        off = (off + nfloats * sizeof(float) + 255) & ~(size_t)255;
        return p;
    };
    WS w;
    w.h    = alloc((size_t)N_ATOM * H);
    w.he   = alloc((size_t)E_ATOM * H);
    w.agg  = alloc((size_t)N_ATOM * H);
    w.big1 = alloc((size_t)N_ATOM * H3);
    w.big2 = alloc((size_t)N_ATOM * H3);
    w.s      = alloc(E_ATOM);
    w.expv   = alloc(E_ATOM);
    w.segmax = (unsigned*)alloc(N_ATOM);
    w.segden = alloc(N_ATOM);
    w.g    = alloc((size_t)N_JUNC * H);
    w.ctx  = alloc((size_t)N_JUNC * H);
    w.xe   = alloc((size_t)N_JUNC * H);
    w.ee   = alloc((size_t)E_JUNC * H);
    w.xcat = alloc((size_t)N_JUNC * 2 * H);
    w.xh   = alloc((size_t)N_JUNC * H);
    w.oc   = alloc((size_t)N_JUNC * 2 * H);
    w.graph_origin = alloc((size_t)NGRAPH * H);
    w.graph_frag   = alloc((size_t)N_JUNC * H);
    w.frag_pool    = alloc((size_t)NGRAPH * H);
    w.superg       = alloc((size_t)NGRAPH * H);
    w.concat3      = alloc((size_t)NGRAPH * H3);
    w.d1 = alloc((size_t)NGRAPH * (H3 / 2));
    w.d2 = alloc((size_t)NGRAPH * (H3 / 2));
    w.colsum = alloc(512); w.colsumsq = alloc(512);

    // ---- origin channel (atom graph) ----
    GraphDesc ga{ x_atom, DIN, ea_atom, EIN, ei_a, ei_a + E_ATOM, b_a, N_ATOM, E_ATOM, NGRAPH };
    for (int hh = 0; hh < 2; ++hh) run_afp(stream, origP.head[hh], ga, w, w.oc + hh * H, 2 * H);
    gemm(stream, w.oc, origP.att_W, origP.att_b, w.ctx, NGRAPH, H, 2 * H, H, nullptr, ACT_NONE, 0);
    bn_block(stream, w, w.ctx, NGRAPH, H, origP.att_g, origP.att_beta, w.graph_origin, ACT_RELU);

    // ---- fragment channel (frag graph -> one embedding per junction node) ----
    GraphDesc gf{ x_frag, DIN, ea_frag, EIN, ei_f, ei_f + E_FRAG, b_f, N_FRAG, E_FRAG, N_JUNC };
    for (int hh = 0; hh < 2; ++hh) run_afp(stream, fragP.head[hh], gf, w, w.oc + hh * H, 2 * H);
    gemm(stream, w.oc, fragP.att_W, fragP.att_b, w.ctx, N_JUNC, H, 2 * H, H, nullptr, ACT_NONE, 0);
    bn_block(stream, w, w.ctx, N_JUNC, H, fragP.att_g, fragP.att_beta, w.graph_frag, ACT_RELU);

    // ---- junction channel ----
    gemm(stream, x_motif, juncP.emb_fW, juncP.emb_fb, w.ctx, N_JUNC, H, FRAG_DIM, H, nullptr, ACT_NONE, 0);
    bn_block(stream, w, w.ctx, N_JUNC, H, juncP.emb_fg, juncP.emb_fbeta, w.xe, ACT_LRELU);
    gemm(stream, ea_junc, juncP.emb_eW, juncP.emb_eb, w.big1, E_JUNC, H, EIN, H, nullptr, ACT_NONE, 0);
    bn_block(stream, w, w.big1, E_JUNC, H, juncP.emb_eg, juncP.emb_ebeta, w.ee, ACT_LRELU);
    k_copy_cols<<<gsb((size_t)N_JUNC * H), 256, 0, stream>>>(w.graph_frag, N_JUNC, H, w.xcat, 2 * H);
    k_copy_cols<<<gsb((size_t)N_JUNC * H), 256, 0, stream>>>(w.xe, N_JUNC, H, w.xcat + H, 2 * H);
    for (int hh = 0; hh < 2; ++hh) {
        gemm(stream, w.xcat, juncP.head[hh].proj_W, juncP.head[hh].proj_b, w.xh,
             N_JUNC, H, 2 * H, H, nullptr, ACT_NONE, 0);
        GraphDesc gj{ w.xh, H, w.ee, H, ei_j, ei_j + E_JUNC, b_j, N_JUNC, E_JUNC, NGRAPH };
        run_afp(stream, juncP.head[hh].afp, gj, w, w.oc + (size_t)hh * NGRAPH * H, H);
    }
    k_mean2relu<<<gsb((size_t)NGRAPH * H), 256, 0, stream>>>(
        w.oc, w.oc + (size_t)NGRAPH * H, w.superg, (size_t)NGRAPH * H);

    // ---- fragment pooling + final concat ----
    fillf(stream, w.frag_pool, 0.f, (size_t)NGRAPH * H);
    k_segsum_feat<<<gsb((size_t)N_JUNC * H), 256, 0, stream>>>(w.graph_frag, b_j, w.frag_pool, N_JUNC);
    k_copy_cols<<<gsb((size_t)NGRAPH * H), 256, 0, stream>>>(w.graph_origin, NGRAPH, H, w.concat3, H3);
    k_copy_cols<<<gsb((size_t)NGRAPH * H), 256, 0, stream>>>(w.frag_pool, NGRAPH, H, w.concat3 + H, H3);
    k_copy_cols<<<gsb((size_t)NGRAPH * H), 256, 0, stream>>>(w.superg, NGRAPH, H, w.concat3 + 2 * H, H3);

    // ---- prediction head ----
    const int HD = H3 / 2;  // 192
    gemm(stream, w.concat3, p1W, p1b, w.d1, NGRAPH, HD, H3, HD, nullptr, ACT_LRELU7, 0);
    bn_block(stream, w, w.d1, NGRAPH, HD, p1g, p1beta, w.d1, ACT_NONE);
    gemm(stream, w.d1, p2W, p2b, w.d2, NGRAPH, HD, HD, HD, nullptr, ACT_LRELU7, 0);
    k_gemv_out<<<DIV_UP(NGRAPH, 8), 256, 0, stream>>>(w.d2, poW, pob, (float*)d_out, NGRAPH, HD);
}
